// LateFusion_15693810500160
// MI455X (gfx1250) — matrix operand order repaired and checked by reading, not compiled, not run on hardware
//
#include <hip/hip_runtime.h>

// CDNA5 / gfx1250, wave32.
typedef __attribute__((ext_vector_type(16))) _Float16 v16h;
typedef __attribute__((ext_vector_type(8)))  float    v8f;

#define BSZ 1024
#define TT  256
#define FF  128
#define HH  16
#define WAVES_PER_BLOCK 8

// ---- fast nonlinearities on the TRANS pipe ----
__device__ __forceinline__ float fast_tanh(float x) {
#if __has_builtin(__builtin_amdgcn_tanhf)
    return __builtin_amdgcn_tanhf(x);          // v_tanh_f32 (CDNA5 trans op)
#else
    float e = __builtin_amdgcn_exp2f(-2.885390082f * x);   // exp(-2x)
    return (1.0f - e) * __builtin_amdgcn_rcpf(1.0f + e);
#endif
}
__device__ __forceinline__ float fast_sig(float x) {
    return 0.5f + 0.5f * fast_tanh(0.5f * x);  // sigmoid via single tanh
}

// Pack hidden state (C/D layout: VGPR=r -> hidden r(+8 for hi half), lane%16=batch)
// into the B-matrix layout of v_wmma_f32_16x16x32_f16:
//   lanes 0-15 : elements 0..7 = K(hidden)=0..7, elements 8..15 = K16..23 (zero)
//   lanes 16-31: elements 0..7 = K(hidden)=8..15, elements 8..15 = K24..31 (zero)
__device__ __forceinline__ v16h pack_B(const float h[8], int lane) {
    v16h b;
    const bool lo = (lane < 16);
#pragma unroll
    for (int r = 0; r < 8; ++r) {
        b[r]     = (_Float16)h[r];
        b[r + 8] = (_Float16)0.0f;
    }
    return b;
}

__global__ void init_out_kernel(float* out, const float* b_fuse) {
    int i = blockIdx.x * blockDim.x + threadIdx.x;
    if (i < BSZ) out[i] = b_fuse[0];
}

__global__ __launch_bounds__(256)
void lstm_fused_kernel(const float* __restrict__ x,       // (B,T,F)
                       const float* __restrict__ W_ih,    // (F,64,1)
                       const float* __restrict__ W_hh,    // (F,64,16)
                       const float* __restrict__ b_ih,    // (F,64)
                       const float* __restrict__ b_hh,    // (F,64)
                       const float* __restrict__ init_h,  // (F,16)
                       const float* __restrict__ init_c,  // (F,16)
                       const float* __restrict__ W_fuse,  // (F*16,1)
                       float* __restrict__ out)           // (B,1)
{
    const int lane  = threadIdx.x & 31;
    const int wave  = threadIdx.x >> 5;
    const int f     = blockIdx.x % FF;                       // feature channel
    const int btile = (blockIdx.x / FF) * WAVES_PER_BLOCK + wave; // 0..63
    const int half  = lane >> 4;       // 0: hidden 0-7 rows, 1: hidden 8-15 rows
    const int nlo   = lane & 15;       // batch within tile / M row within tile
    const int batch = btile * 16 + nlo;

    // ---- A tiles: W_hh[f] as f16, M=gate row, K=hidden (padded 16->32) ----
    // A layout: lanes 0-15 (M=lane): elems 0..7 = K0..7, elems 8..15 = K16..23 (zero)
    //           lanes 16-31 (M=lane-16): elems 0..7 = K8..15, elems 8..15 = K24..31 (zero)
    v16h A[4];
#pragma unroll
    for (int q = 0; q < 4; ++q) {
        const float* wp = W_hh + ((size_t)(f * 64 + q * 16 + nlo)) * 16 + half * 8;
#pragma unroll
        for (int k = 0; k < 8; ++k)  A[q][k] = (_Float16)wp[k];
#pragma unroll
        for (int k = 8; k < 16; ++k) A[q][k] = (_Float16)0.0f;
    }

    // ---- per-lane gate constants in C/D layout: reg r -> gate q*16 + r + 8*half ----
    float wx[4][8], bb[4][8];
#pragma unroll
    for (int q = 0; q < 4; ++q) {
#pragma unroll
        for (int r = 0; r < 8; ++r) {
            int g = q * 16 + 8 * half + r;
            wx[q][r] = W_ih[f * 64 + g];
            bb[q][r] = b_ih[f * 64 + g] + b_hh[f * 64 + g];
        }
    }

    // ---- state in C/D layout ----
    float c[8], h[8], wf[8];
#pragma unroll
    for (int r = 0; r < 8; ++r) {
        int hid = 8 * half + r;
        c[r]  = init_c[f * HH + hid];
        h[r]  = init_h[f * HH + hid];
        wf[r] = W_fuse[f * HH + hid];
    }
    v16h Bh = pack_B(h, lane);

    // ---- recurrence over T steps ----
    const float* xp = x + (size_t)batch * (TT * FF) + f;
    float xv = xp[0];                       // software-pipelined scalar input
    for (int t = 0; t < TT; ++t) {
        __builtin_prefetch(xp + 8 * FF, 0, 1);        // global_prefetch_b8 (gfx1250)
        float xv_next = (t + 1 < TT) ? xp[FF] : 0.0f; // next step's input
        xp += FF;

        v8f d[4];
#pragma unroll
        for (int q = 0; q < 4; ++q) {
            v8f cacc;
#pragma unroll
            for (int r = 0; r < 8; ++r) cacc[r] = fmaf(xv, wx[q][r], bb[q][r]);
            // D = W_hh_tile(A) x h^T(B) + (x*Wx + bias)
            d[q] = __builtin_amdgcn_wmma_f32_16x16x32_f16(
                /*neg_a=*/false, A[q], /*neg_b=*/false, Bh,
                /*c_mod=*/(short)0, cacc, /*reuse_a=*/false, /*reuse_b=*/false);
        }

#pragma unroll
        for (int r = 0; r < 8; ++r) {
            float ig = fast_sig(d[0][r]);
            float fg = fast_sig(d[1][r]);
            float gg = fast_tanh(d[2][r]);
            float og = fast_sig(d[3][r]);
            c[r] = fg * c[r] + ig * gg;
            h[r] = og * fast_tanh(c[r]);
        }
        Bh = pack_B(h, lane);
        xv = xv_next;
    }

    // ---- fuse: out[b] += sum_hid h[b, f, hid] * W_fuse[f*16+hid] ----
    float s = 0.0f;
#pragma unroll
    for (int r = 0; r < 8; ++r) s = fmaf(h[r], wf[r], s);
    s += __shfl_xor(s, 16, 32);             // combine hidden 0-7 and 8-15 halves
    if (lane < 16) atomicAdd(out + btile * 16 + lane, s);
}

extern "C" void kernel_launch(void* const* d_in, const int* in_sizes, int n_in,
                              void* d_out, int out_size, void* d_ws, size_t ws_size,
                              hipStream_t stream) {
    const float* x      = (const float*)d_in[0];
    const float* W_ih   = (const float*)d_in[1];
    const float* W_hh   = (const float*)d_in[2];
    const float* b_ih   = (const float*)d_in[3];
    const float* b_hh   = (const float*)d_in[4];
    const float* init_h = (const float*)d_in[5];
    const float* init_c = (const float*)d_in[6];
    const float* W_fuse = (const float*)d_in[7];
    const float* b_fuse = (const float*)d_in[8];
    float* out = (float*)d_out;

    init_out_kernel<<<(BSZ + 255) / 256, 256, 0, stream>>>(out, b_fuse);

    // grid: F features x (64 batch-tiles / 8 waves-per-block) = 1024 blocks, 8 waves each
    const int nblocks = FF * ((BSZ / 16) / WAVES_PER_BLOCK);
    lstm_fused_kernel<<<nblocks, 32 * WAVES_PER_BLOCK, 0, stream>>>(
        x, W_ih, W_hh, b_ih, b_hh, init_h, init_c, W_fuse, out);
}
